// HopfRNNLayerTheta_80221399154877
// MI455X (gfx1250) — matrix operand order repaired and checked by reading, not compiled, not run on hardware
//
#include <hip/hip_runtime.h>
#include <math.h>

#define NFFT 3073
#define BSZ  64
#define TLEN 2048
#define UNITS 1024
#define PI_D 3.14159265358979323846

typedef __attribute__((ext_vector_type(16))) _Float16 v16h;
typedef __attribute__((ext_vector_type(8)))  float    v8f;
typedef __attribute__((ext_vector_type(4)))  float    v4f;

struct cf { float x, y; };
__device__ __forceinline__ cf cmul(cf a, cf b){ return cf{a.x*b.x - a.y*b.y, a.x*b.y + a.y*b.x}; }
__device__ __forceinline__ cf cadd(cf a, cf b){ return cf{a.x+b.x, a.y+b.y}; }
__device__ __forceinline__ cf csub(cf a, cf b){ return cf{a.x-b.x, a.y-b.y}; }

// ------------------------------------------------------------------
// Kernel 1: X[j,t] = rfft(x[j], n=3073)[t+1], t = 0..1023 (direct DFT,
// phasor recurrence refreshed every 256 steps). 1 thread per (j,t);
// all lanes of a wave share j -> broadcast loads of x.
// ------------------------------------------------------------------
__global__ __launch_bounds__(256) void dft_kernel(const float* __restrict__ x,
                                                  float2* __restrict__ Xbuf)
{
  int tid = blockIdx.x * blockDim.x + threadIdx.x;   // 65536 total
  int j = tid >> 10;
  int t = tid & (UNITS - 1);
  int k = t + 1;
  double w = -2.0 * PI_D * (double)k / (double)NFFT;
  float sw, cw;
  __sincosf((float)w, &sw, &cw);
  cf wstep{cw, sw};
  cf p{1.f, 0.f};
  cf acc{0.f, 0.f};
  const float* xr = x + j * TLEN;
  for (int n = 0; n < TLEN; ++n) {
    if ((n & 255) == 0) {                    // drift control
      float ang = (float)fmod(w * (double)n, 2.0 * PI_D);
      float s, c; __sincosf(ang, &s, &c);
      p = cf{c, s};
    }
    float xv = xr[n];
    acc.x = fmaf(xv, p.x, acc.x);
    acc.y = fmaf(xv, p.y, acc.y);
    p = cmul(p, wstep);
  }
  Xbuf[j * UNITS + t] = make_float2(acc.x, acc.y);
}

// ------------------------------------------------------------------
// Kernel 2: g[b][m] = z0[b][m+1] for m<=1022; zero the pad slot m=2047.
// ------------------------------------------------------------------
__global__ __launch_bounds__(256) void prep_kernel(const float2* __restrict__ z0,
                                                   float2* __restrict__ gbuf)
{
  int b = blockIdx.x;
  for (int m = threadIdx.x; m < 2048; m += blockDim.x) {
    if (m <= 1022)      gbuf[b * 2048 + m] = z0[b * UNITS + m + 1];
    else if (m == 2047) gbuf[b * 2048 + m] = make_float2(0.f, 0.f);
    // m in [1023,2046] written by the WMMA GEMM kernel
  }
}

// ------------------------------------------------------------------
// Kernel 3 (WMMA): y = 0.5*(A @ Z0head + Bw @ X), complex 64x64 x 64x1024.
// One wave per 16x16 output tile; v_wmma_f32_16x16x32_f16, K=64 in 2 steps.
// Complex mul via 3 accumulators: accR (+re), accRm (re to subtract), accI.
// Result stored into g[b][1023+t].
// ------------------------------------------------------------------
__global__ __launch_bounds__(128) void gemm_kernel(const float2* __restrict__ Amat,
                                                   const float2* __restrict__ Bw,
                                                   const float2* __restrict__ z0,
                                                   const float2* __restrict__ Xbuf,
                                                   float2* __restrict__ gbuf)
{
  int tile = blockIdx.x * 4 + (threadIdx.x >> 5);   // 256 tiles: 4 (b) x 64 (t)
  int lane = threadIdx.x & 31;
  int bt = (tile & 3) << 4;
  int tt = (tile >> 2) << 4;
  int lrow = lane & 15;
  int lhi  = lane >> 4;
  v8f accR = {}; v8f accRm = {}; v8f accI = {};
  #pragma unroll
  for (int ks = 0; ks < 2; ++ks) {
    int k0 = ks << 5;
    v16h bwR, bwI, aR, aI, xR, xI, zR, zI;
    #pragma unroll
    for (int e = 0; e < 16; ++e) {
      // A-fragment (16x32 f16): lane holds row M=lane&15; K per ISA layout
      int K = k0 + ((e >> 3) << 4) + (lhi << 3) + (e & 7);
      float2 bw = Bw  [(bt + lrow) * BSZ + K];
      float2 av = Amat[(bt + lrow) * BSZ + K];
      bwR[e] = (_Float16)bw.x; bwI[e] = (_Float16)bw.y;
      aR[e]  = (_Float16)av.x; aI[e]  = (_Float16)av.y;
      // B-fragment (32x16 f16): col N=lane&15, K = k0 + (e>>3)*16 + (lane>>4)*8 + (e&7)
      int Kb = k0 + ((e >> 3) << 4) + (lhi << 3) + (e & 7);
      float2 xv = Xbuf[Kb * UNITS + tt + lrow];
      float2 zv = z0  [Kb * UNITS + tt + lrow];
      xR[e] = (_Float16)xv.x; xI[e] = (_Float16)xv.y;
      zR[e] = (_Float16)zv.x; zI[e] = (_Float16)zv.y;
    }
    accR  = __builtin_amdgcn_wmma_f32_16x16x32_f16(false, bwR, false, xR, (short)0, accR,  false, false);
    accR  = __builtin_amdgcn_wmma_f32_16x16x32_f16(false, aR,  false, zR, (short)0, accR,  false, false);
    accRm = __builtin_amdgcn_wmma_f32_16x16x32_f16(false, bwI, false, xI, (short)0, accRm, false, false);
    accRm = __builtin_amdgcn_wmma_f32_16x16x32_f16(false, aI,  false, zI, (short)0, accRm, false, false);
    accI  = __builtin_amdgcn_wmma_f32_16x16x32_f16(false, bwR, false, xI, (short)0, accI,  false, false);
    accI  = __builtin_amdgcn_wmma_f32_16x16x32_f16(false, bwI, false, xR, (short)0, accI,  false, false);
    accI  = __builtin_amdgcn_wmma_f32_16x16x32_f16(false, aR,  false, zI, (short)0, accI,  false, false);
    accI  = __builtin_amdgcn_wmma_f32_16x16x32_f16(false, aI,  false, zR, (short)0, accI,  false, false);
  }
  #pragma unroll
  for (int v = 0; v < 8; ++v) {                      // C/D layout: M=(lane>>4)*8+v
    int row = bt + (lhi << 3) + v;
    int col = tt + lrow;
    float2 g;
    g.x = 0.5f * (accR[v] - accRm[v]);
    g.y = 0.5f * accI[v];
    gbuf[row * 2048 + 1023 + col] = g;
  }
}

// ------------------------------------------------------------------
// Kernel 4: bandwidth-bound sliding-window DFT.
// out[b,s,u] = (1/N)*(2*Re(conj(phi^u) * S[u]) - Re(g[u])),
//   S[u] = PreExc[u+1024] - PreExc[u],  T[m] = g[m]*phi^m.
// One wave per (b,s); lanes cover 4 consecutive u's; wave prefix scans.
// All 8 waves of a block share batch b: stage the 16KB g-row into LDS
// once per block with CDNA5 async copy (ASYNCcnt), then read from LDS.
// Output written with non-temporal stores (512MB, write-once).
// ------------------------------------------------------------------
__device__ __forceinline__ cf wave_incl_scan(cf v, int lane){
  #pragma unroll
  for (int d = 1; d < 32; d <<= 1) {
    float tx = __shfl_up(v.x, (unsigned)d, 32);
    float ty = __shfl_up(v.y, (unsigned)d, 32);
    if (lane >= d) { v.x += tx; v.y += ty; }
  }
  return v;
}

__global__ __launch_bounds__(256) void irfft_kernel(const float2* __restrict__ gbuf,
                                                    float* __restrict__ out)
{
  __shared__ float2 gs[2048];                        // one 16KB g-row per block
  int w = blockIdx.x * 8 + (threadIdx.x >> 5);       // 131072 waves
  int lane = threadIdx.x & 31;
  int b = blockIdx.x >> 8;                           // 256 blocks per batch row
  int s = w & 2047;

  // ---- async stage gbuf row -> LDS (global_load_async_to_lds_b128) ----
  {
    unsigned ldsbase = (unsigned)(unsigned long long)(void*)&gs[0];
    unsigned long long gbase = (unsigned long long)(gbuf + (size_t)b * 2048);
    for (int i = threadIdx.x; i < 1024; i += 256) {  // 1024 x 16B = 16KB
      unsigned dst = ldsbase + (unsigned)(i << 4);
      unsigned long long src = gbase + ((unsigned long long)i << 4);
      asm volatile("global_load_async_to_lds_b128 %0, %1, off"
                   :: "v"(dst), "v"(src) : "memory");
    }
    asm volatile("s_wait_asynccnt 0" ::: "memory");
    __syncthreads();
  }

  const float invN = 1.0f / (float)NFFT;
  const double thd = 2.0 * PI_D * (double)s / (double)NFFT;
  float s1, c1; __sincosf((float)thd, &s1, &c1);
  cf phi1{c1, s1};
  float aK = (float)fmod(thd * 1024.0, 2.0 * PI_D);
  float sk, ck; __sincosf(aK, &sk, &ck);
  cf K1024{ck, sk};                                  // phi^1024
  float* orow = out + ((size_t)b * 2048 + (size_t)s) * 1024;

  // pre-pass: carryB = sum_{m=0}^{1023} g[m]*phi^m  (PreExc[1024])
  cf carryB{0.f, 0.f};
  {
    float ang0 = (float)fmod(thd * (double)lane, 2.0 * PI_D);
    float ss, cc; __sincosf(ang0, &ss, &cc);
    cf p{cc, ss};
    float a32 = (float)fmod(thd * 32.0, 2.0 * PI_D);
    float s32, c32; __sincosf(a32, &s32, &c32);
    cf step32{c32, s32};
    cf tot{0.f, 0.f};
    for (int i = 0; i < 32; ++i) {
      float2 g = gs[lane + (i << 5)];
      tot = cadd(tot, cmul(cf{g.x, g.y}, p));
      p = cmul(p, step32);
    }
    #pragma unroll
    for (int d = 1; d < 32; d <<= 1) {
      tot.x += __shfl_xor(tot.x, d, 32);
      tot.y += __shfl_xor(tot.y, d, 32);
    }
    carryB = tot;
  }
  cf carryA{0.f, 0.f};

  for (int c = 0; c < 8; ++c) {
    int m0 = (c << 7) + (lane << 2);                 // u base for this lane
    float ang = (float)fmod(thd * (double)m0, 2.0 * PI_D);
    float ss, cc; __sincosf(ang, &ss, &cc);
    cf p0{cc, ss};
    cf p1 = cmul(p0, phi1);
    cf p2 = cmul(p1, phi1);
    cf p3 = cmul(p2, phi1);
    cf q0 = cmul(p0, K1024);                         // phi^(m0+1024+j)
    cf q1 = cmul(p1, K1024);
    cf q2 = cmul(p2, K1024);
    cf q3 = cmul(p3, K1024);
    float2 gA0 = gs[m0],      gA1 = gs[m0+1],    gA2 = gs[m0+2],    gA3 = gs[m0+3];
    float2 gB0 = gs[m0+1024], gB1 = gs[m0+1025], gB2 = gs[m0+1026], gB3 = gs[m0+1027];
    cf sA0 = cmul(cf{gA0.x, gA0.y}, p0);
    cf sA1 = cadd(sA0, cmul(cf{gA1.x, gA1.y}, p1));
    cf sA2 = cadd(sA1, cmul(cf{gA2.x, gA2.y}, p2));
    cf sA3 = cadd(sA2, cmul(cf{gA3.x, gA3.y}, p3));
    cf sB0 = cmul(cf{gB0.x, gB0.y}, q0);
    cf sB1 = cadd(sB0, cmul(cf{gB1.x, gB1.y}, q1));
    cf sB2 = cadd(sB1, cmul(cf{gB2.x, gB2.y}, q2));
    cf sB3 = cadd(sB2, cmul(cf{gB3.x, gB3.y}, q3));
    cf inA = wave_incl_scan(sA3, lane);
    cf inB = wave_incl_scan(sB3, lane);
    cf baseA{carryA.x + inA.x - sA3.x, carryA.y + inA.y - sA3.y};  // PreExc at lane start
    cf baseB{carryB.x + inB.x - sB3.x, carryB.y + inB.y - sB3.y};
    float o0, o1, o2, o3;
    { cf S = csub(baseB, baseA);
      o0 = invN * (2.f * (p0.x * S.x + p0.y * S.y) - gA0.x); }
    { cf S = csub(cadd(baseB, sB0), cadd(baseA, sA0));
      o1 = invN * (2.f * (p1.x * S.x + p1.y * S.y) - gA1.x); }
    { cf S = csub(cadd(baseB, sB1), cadd(baseA, sA1));
      o2 = invN * (2.f * (p2.x * S.x + p2.y * S.y) - gA2.x); }
    { cf S = csub(cadd(baseB, sB2), cadd(baseA, sA2));
      o3 = invN * (2.f * (p3.x * S.x + p3.y * S.y) - gA3.x); }
    v4f ov = { o0, o1, o2, o3 };
    __builtin_nontemporal_store(ov, reinterpret_cast<v4f*>(orow + m0));
    float tAx = __shfl(inA.x, 31, 32), tAy = __shfl(inA.y, 31, 32);
    float tBx = __shfl(inB.x, 31, 32), tBy = __shfl(inB.y, 31, 32);
    carryA.x += tAx; carryA.y += tAy;
    carryB.x += tBx; carryB.y += tBy;
  }
}

// ------------------------------------------------------------------
extern "C" void kernel_launch(void* const* d_in, const int* in_sizes, int n_in,
                              void* d_out, int out_size, void* d_ws, size_t ws_size,
                              hipStream_t stream)
{
  (void)in_sizes; (void)n_in; (void)out_size; (void)ws_size;
  const float*  x    = (const float*) d_in[0];   // (64,2048,1) f32
  const float2* Amat = (const float2*)d_in[1];   // (64,64) c64
  const float2* Bw   = (const float2*)d_in[2];   // (64,64) c64
  const float2* z0   = (const float2*)d_in[3];   // (64,1024) c64
  float* out = (float*)d_out;                    // (64,2048,1024) f32

  float2* Xbuf = (float2*)d_ws;                                             // 512 KB
  float2* gbuf = (float2*)((char*)d_ws + (size_t)BSZ * UNITS * sizeof(float2)); // 1 MB

  dft_kernel <<<256,   256, 0, stream>>>(x, Xbuf);
  prep_kernel<<<BSZ,   256, 0, stream>>>(z0, gbuf);
  gemm_kernel<<<64,    128, 0, stream>>>(Amat, Bw, z0, Xbuf, gbuf);
  irfft_kernel<<<16384, 256, 0, stream>>>(gbuf, out);
}
